// GroundedMambaBlock_16999480558028
// MI455X (gfx1250) — matrix operand order repaired and checked by reading, not compiled, not run on hardware
//
#include <hip/hip_runtime.h>
#include <hip/hip_bf16.h>

// ---------------- problem constants ----------------
#define Bsz 4
#define Lsz 2048
#define Dm  1024
#define Hn  16
#define Sd  16
#define HD  64
#define Mrows (Bsz*Lsz)          // 8192

typedef __attribute__((ext_vector_type(16))) __bf16 v16bf;
typedef __attribute__((ext_vector_type(8)))  __bf16 v8bf;
typedef __attribute__((ext_vector_type(8)))  float  v8f;

static __device__ __forceinline__ unsigned short f2bf(float f) {
    union { float f; unsigned u; } un; un.f = f;
    unsigned r = un.u + 0x7FFFu + ((un.u >> 16) & 1u);   // RNE
    return (unsigned short)(r >> 16);
}
static __device__ __forceinline__ float bf2f(unsigned short h) {
    union { unsigned u; float f; } un; un.u = ((unsigned)h) << 16;
    return un.f;
}
static __device__ __forceinline__ v16bf make16(v8bf lo, v8bf hi) {
    v16bf r;
#pragma unroll
    for (int i = 0; i < 8; ++i) { r[i] = lo[i]; r[i + 8] = hi[i]; }
    return r;
}

// ---------------- 1) dual LayerNorm: xs (f32) + xr (bf16) ----------------
__global__ void ln_dual_kernel(const float* __restrict__ x,
                               const float* __restrict__ gs, const float* __restrict__ bs,
                               const float* __restrict__ gr, const float* __restrict__ br,
                               float* __restrict__ xs, unsigned short* __restrict__ xr) {
    const int row = blockIdx.x;
    const float* xp = x + (size_t)row * Dm;
    float v[4]; float s = 0.f, sq = 0.f;
#pragma unroll
    for (int i = 0; i < 4; ++i) {
        int d = threadIdx.x + i * 256;
        v[i] = xp[d]; s += v[i]; sq += v[i] * v[i];
    }
    __shared__ float rs[256], rq[256];
    rs[threadIdx.x] = s; rq[threadIdx.x] = sq; __syncthreads();
    for (int o = 128; o > 0; o >>= 1) {
        if (threadIdx.x < o) { rs[threadIdx.x] += rs[threadIdx.x + o]; rq[threadIdx.x] += rq[threadIdx.x + o]; }
        __syncthreads();
    }
    const float mean = rs[0] * (1.f / Dm);
    const float var  = rq[0] * (1.f / Dm) - mean * mean;
    const float rstd = rsqrtf(var + 1e-5f);
#pragma unroll
    for (int i = 0; i < 4; ++i) {
        int d = threadIdx.x + i * 256;
        float n = (v[i] - mean) * rstd;
        xs[(size_t)row * Dm + d] = n * gs[d] + bs[d];
        xr[(size_t)row * Dm + d] = f2bf(n * gr[d] + br[d]);
    }
}

// ---------------- 2) weight transpose + bf16: Wt[k][n] = W[n][k] ----------------
__global__ void transpose_bf16_kernel(const float* __restrict__ W, unsigned short* __restrict__ Wt) {
    __shared__ float tile[16][17];
    const int bx = blockIdx.x * 16, by = blockIdx.y * 16;
    const int tx = threadIdx.x % 16, ty = threadIdx.x / 16;
    tile[ty][tx] = W[(size_t)(by + ty) * Dm + (bx + tx)];
    __syncthreads();
    Wt[(size_t)(bx + ty) * Dm + (by + tx)] = f2bf(tile[tx][ty]);
}

// ---------------- 3) spectral norm of A (power iteration), An = A*scale ----------------
__global__ void spectral_an_kernel(const float* __restrict__ A, float* __restrict__ An) {
    __shared__ float G[16][16];
    __shared__ float vv[16], wv[16];
    __shared__ float lam, scale;
    const int i = threadIdx.x / 16, j = threadIdx.x % 16;
    float g = 0.f;
#pragma unroll
    for (int k = 0; k < 16; ++k) g += A[k * 16 + i] * A[k * 16 + j];   // G = A^T A
    G[i][j] = g;
    if (threadIdx.x < 16) vv[threadIdx.x] = 1.f;
    __syncthreads();
    for (int it = 0; it < 64; ++it) {
        if (threadIdx.x < 16) {
            float w = 0.f;
#pragma unroll
            for (int jj = 0; jj < 16; ++jj) w += G[threadIdx.x][jj] * vv[jj];
            wv[threadIdx.x] = w;
        }
        __syncthreads();
        if (threadIdx.x == 0) {
            float n = 0.f;
            for (int k = 0; k < 16; ++k) n += wv[k] * wv[k];
            n = sqrtf(n); lam = n;
            float inv = (n > 0.f) ? 1.f / n : 0.f;
            for (int k = 0; k < 16; ++k) vv[k] = wv[k] * inv;
        }
        __syncthreads();
    }
    if (threadIdx.x == 0) {
        float s0 = sqrtf(lam);
        scale = (s0 > 0.99f) ? (0.99f / s0) : 1.f;
    }
    __syncthreads();
    An[i * 16 + j] = A[i * 16 + j] * scale;
}

// ---------------- 4) dt and Bx: per-token, wave per output ----------------
__global__ void dtbx_kernel(const float* __restrict__ xs, const float* __restrict__ B_w,
                            const float* __restrict__ dt_w, const float* __restrict__ dt_b,
                            float* __restrict__ dt, float* __restrict__ Bx) {
    const int row = blockIdx.x;
    const int wave = threadIdx.x >> 5, lane = threadIdx.x & 31;
    const float* xp = xs + (size_t)row * Dm;
    for (int o = wave; o < 17; o += 8) {
        const float* wp = (o < 16) ? (B_w + (size_t)o * Dm) : dt_w;
        float acc = 0.f;
        for (int d = lane; d < Dm; d += 32) acc += xp[d] * wp[d];
#pragma unroll
        for (int m = 16; m >= 1; m >>= 1) acc += __shfl_xor(acc, m, 32);
        if (lane == 0) {
            if (o < 16) Bx[(size_t)row * Sd + o] = acc;
            else {
                float z = acc + dt_b[0];
                float sp = (z > 20.f) ? z : log1pf(expf(z));
                dt[row] = fminf(0.1f, fmaxf(1e-3f, sp));
            }
        }
    }
}

// ---------------- 5) SSM sequential scan (1 block, wave per batch) ----------------
__global__ void ssm_scan_kernel(const float* __restrict__ An, const float* __restrict__ dt,
                                const float* __restrict__ Bx,
                                const float* __restrict__ sn_g, const float* __restrict__ sn_b,
                                float* __restrict__ states, float* __restrict__ ssm_state_out) {
    const int b = threadIdx.x >> 5, lane = threadIdx.x & 31;
    const bool act = lane < 16;
    float Arow[16];
#pragma unroll
    for (int j = 0; j < 16; ++j) Arow[j] = act ? An[lane * 16 + j] : 0.f;
    const float g  = act ? sn_g[lane] : 1.f;
    const float be = act ? sn_b[lane] : 0.f;
    float st = 0.f;
    for (int t = 0; t < Lsz; ++t) {
        const float dtt = dt[b * Lsz + t];
        const float bx = act ? Bx[(size_t)(b * Lsz + t) * Sd + lane] : 0.f;
        float contrib = 0.f;
#pragma unroll
        for (int j = 0; j < 16; ++j) contrib += Arow[j] * __shfl(st, j, 16);
        st = st + dtt * contrib + bx;
        // LayerNorm over the 16 lanes
        float s = st;
#pragma unroll
        for (int m = 8; m >= 1; m >>= 1) s += __shfl_xor(s, m, 16);
        const float mean = s * (1.f / 16.f);
        float dsq = (st - mean) * (st - mean);
#pragma unroll
        for (int m = 8; m >= 1; m >>= 1) dsq += __shfl_xor(dsq, m, 16);
        const float var = dsq * (1.f / 16.f);
        st = (st - mean) * rsqrtf(var + 1e-5f) * g + be;
        if (act) states[(size_t)(b * Lsz + t) * Sd + lane] = st;
    }
    if (act) ssm_state_out[b * Sd + lane] = st;
}

// ---------------- 6) ssm_out = states @ C_w^T + D_vec * xs  (in-place over xs) ----------------
__global__ void ssm_out_kernel(const float* __restrict__ states, const float* __restrict__ C_w,
                               const float* __restrict__ D_vec, float* __restrict__ xs_io) {
    const int row = blockIdx.x;
    __shared__ float st[16];
    if (threadIdx.x < 16) st[threadIdx.x] = states[(size_t)row * Sd + threadIdx.x];
    __syncthreads();
#pragma unroll
    for (int i = 0; i < 4; ++i) {
        int d = threadIdx.x + i * 256;
        float acc = D_vec[d] * xs_io[(size_t)row * Dm + d];
        const float* c = C_w + (size_t)d * Sd;
#pragma unroll
        for (int s = 0; s < 16; ++s) acc += st[s] * c[s];
        xs_io[(size_t)row * Dm + d] = acc;
    }
}

// ---------------- 7) WMMA bf16 GEMM: C[M,N] = A[M,K] * Bt[K,N] ----------------
// Bt is pre-transposed weights (W^T, K-major). One wave -> 16x64 output strip.
__global__ void gemm_bf16_wmma_kernel(const unsigned short* __restrict__ Abf,
                                      const unsigned short* __restrict__ Bt,
                                      void* __restrict__ Cout,
                                      int M, int N, int K, int out_bf16) {
    const int wave = threadIdx.x >> 5;
    const int lane = threadIdx.x & 31;
    const int mtile = blockIdx.x * 16;
    const int nbase = blockIdx.y * 512 + wave * 64;
    const int h = lane >> 4;                       // lane group (K-half select)
    const unsigned short* Arow = Abf + (size_t)(mtile + (lane & 15)) * K;

    v8f acc0 = {}, acc1 = {}, acc2 = {}, acc3 = {};
    for (int k0 = 0; k0 < K; k0 += 32) {
        // A fragment: M = lane%16, K(e) = (e<8 ? e : e+8) + 8*h  within [k0, k0+32)
        const v8bf* ap = (const v8bf*)(Arow + k0 + 8 * h);
        v16bf va = make16(ap[0], ap[2]);
        __builtin_prefetch(Arow + k0 + 64, 0, 1);
        // B fragment: N = lane%16 (+16*tile), K(e) = (e<8 ? e : e+8) + 8*h  within [k0, k0+32)
        const unsigned short* bp = Bt + (size_t)(k0 + 8 * h) * N + nbase + (lane & 15);
        v16bf vb0, vb1, vb2, vb3;
#pragma unroll
        for (int e = 0; e < 16; ++e) {
            const size_t ko = (size_t)((e < 8) ? e : e + 8) * N;
            vb0[e] = __builtin_bit_cast(__bf16, bp[ko]);
            vb1[e] = __builtin_bit_cast(__bf16, bp[ko + 16]);
            vb2[e] = __builtin_bit_cast(__bf16, bp[ko + 32]);
            vb3[e] = __builtin_bit_cast(__bf16, bp[ko + 48]);
        }
        acc0 = __builtin_amdgcn_wmma_f32_16x16x32_bf16(false, va, false, vb0, (short)0, acc0, false, false);
        acc1 = __builtin_amdgcn_wmma_f32_16x16x32_bf16(false, va, false, vb1, (short)0, acc1, false, false);
        acc2 = __builtin_amdgcn_wmma_f32_16x16x32_bf16(false, va, false, vb2, (short)0, acc2, false, false);
        acc3 = __builtin_amdgcn_wmma_f32_16x16x32_bf16(false, va, false, vb3, (short)0, acc3, false, false);
    }
    // C/D layout: VGPR j -> row (j + 8*h), col = lane%16 within each 16-col tile
    const int cbase = nbase + (lane & 15);
#pragma unroll
    for (int t = 0; t < 4; ++t) {
        v8f acc = (t == 0) ? acc0 : (t == 1) ? acc1 : (t == 2) ? acc2 : acc3;
        const int cn = cbase + t * 16;
#pragma unroll
        for (int j = 0; j < 8; ++j) {
            size_t idx = (size_t)(mtile + j + 8 * h) * N + cn;
            if (out_bf16) ((unsigned short*)Cout)[idx] = f2bf(acc[j]);
            else          ((float*)Cout)[idx] = acc[j];
        }
    }
}

// ---------------- 8) RWKV sequential scan: block per (b,h), 64x64 state in regs ----------------
__global__ void rwkv_scan_kernel(const unsigned short* __restrict__ kbf,
                                 const unsigned short* __restrict__ vbf,
                                 const unsigned short* __restrict__ rbf,
                                 const float* __restrict__ time_decay,
                                 const float* __restrict__ time_first,
                                 unsigned short* __restrict__ ys,
                                 float* __restrict__ wkv_out) {
    const int bh = blockIdx.x;
    const int b = bh >> 4, hh = bh & 15;
    const int e = threadIdx.x & 63;
    const int dg = threadIdx.x >> 6;      // 4 groups of 16 d-values
    __shared__ float kk[64], vv[64], rr[64];
    __shared__ float yp[4][64];
    float s[16], dec[16], tf[16];
#pragma unroll
    for (int i = 0; i < 16; ++i) {
        const int d = dg * 16 + i;
        dec[i] = time_decay[hh * HD + d];
        tf[i]  = time_first[hh * HD + d];
        s[i] = 0.f;
    }
    for (int t = 0; t < Lsz; ++t) {
        const size_t base = ((size_t)b * Lsz + t) * Dm + hh * HD;
        if (threadIdx.x < 192) {
            const int j = threadIdx.x & 63, which = threadIdx.x >> 6;
            if (which == 0)      kk[j] = bf2f(kbf[base + j]);
            else if (which == 1) vv[j] = bf2f(vbf[base + j]);
            else { float z = bf2f(rbf[base + j]); rr[j] = 1.f / (1.f + expf(-z)); }
        }
        __syncthreads();
        const float ve = vv[e];
        float y = 0.f;
#pragma unroll
        for (int i = 0; i < 16; ++i) {
            const int d = dg * 16 + i;
            s[i] = s[i] * dec[i] + kk[d] * ve * tf[i];
            y += s[i] * rr[d];
        }
        yp[dg][e] = y;
        __syncthreads();
        if (dg == 0) ys[base + e] = f2bf(yp[0][e] + yp[1][e] + yp[2][e] + yp[3][e]);
        __syncthreads();
    }
#pragma unroll
    for (int i = 0; i < 16; ++i) {
        const int d = dg * 16 + i;
        wkv_out[((size_t)bh * HD + d) * HD + e] = s[i];
    }
}

// ---------------- 9) combine: y = x + LN(0.7*ssm + 0.3*rwkv) ----------------
__global__ void combine_kernel(const float* __restrict__ x, const float* __restrict__ ssm_o,
                               const float* __restrict__ rwkv_o,
                               const float* __restrict__ g, const float* __restrict__ bb,
                               float* __restrict__ y) {
    const int row = blockIdx.x;
    __shared__ float comb[Dm];
    __shared__ float rs[256], rq[256];
    float s = 0.f, sq = 0.f;
#pragma unroll
    for (int i = 0; i < 4; ++i) {
        int d = threadIdx.x + i * 256;
        float c = 0.7f * ssm_o[(size_t)row * Dm + d] + 0.3f * rwkv_o[(size_t)row * Dm + d];
        comb[d] = c; s += c; sq += c * c;
    }
    rs[threadIdx.x] = s; rq[threadIdx.x] = sq; __syncthreads();
    for (int o = 128; o > 0; o >>= 1) {
        if (threadIdx.x < o) { rs[threadIdx.x] += rs[threadIdx.x + o]; rq[threadIdx.x] += rq[threadIdx.x + o]; }
        __syncthreads();
    }
    const float mean = rs[0] * (1.f / Dm);
    const float var  = rq[0] * (1.f / Dm) - mean * mean;
    const float rstd = rsqrtf(var + 1e-5f);
#pragma unroll
    for (int i = 0; i < 4; ++i) {
        int d = threadIdx.x + i * 256;
        y[(size_t)row * Dm + d] = x[(size_t)row * Dm + d] + (comb[d] - mean) * rstd * g[d] + bb[d];
    }
}

// ---------------- host side ----------------
extern "C" void kernel_launch(void* const* d_in, const int* in_sizes, int n_in,
                              void* d_out, int out_size, void* d_ws, size_t ws_size,
                              hipStream_t stream) {
    const float* x   = (const float*)d_in[0];
    const float* wk  = (const float*)d_in[1];
    const float* wv  = (const float*)d_in[2];
    const float* wr  = (const float*)d_in[3];
    const float* wo  = (const float*)d_in[4];
    const float* time_decay = (const float*)d_in[5];
    const float* time_first = (const float*)d_in[6];
    const float* A    = (const float*)d_in[7];
    const float* B_w  = (const float*)d_in[8];
    const float* C_w  = (const float*)d_in[9];
    const float* D_vec= (const float*)d_in[10];
    const float* dt_w = (const float*)d_in[11];
    const float* dt_b = (const float*)d_in[12];
    const float* sn_g = (const float*)d_in[13];
    const float* sn_b = (const float*)d_in[14];
    const float* g_ssm = (const float*)d_in[15];
    const float* b_ssm = (const float*)d_in[16];
    const float* g_rwkv = (const float*)d_in[17];
    const float* b_rwkv = (const float*)d_in[18];
    const float* g_out = (const float*)d_in[19];
    const float* b_out = (const float*)d_in[20];

    // ---- workspace layout (bytes) ----
    char* ws = (char*)d_ws;
    const size_t szXS  = (size_t)Mrows * Dm * 4;   // f32, becomes ssm_out in place
    const size_t szXR  = (size_t)Mrows * Dm * 2;   // bf16
    const size_t szWT  = (size_t)Dm * Dm * 2;      // bf16 transposed weight
    size_t off = 0;
    float*          xs    = (float*)(ws + off);           off += szXS;
    unsigned short* xr    = (unsigned short*)(ws + off);  off += szXR;
    unsigned short* wkT   = (unsigned short*)(ws + off);  off += szWT;
    unsigned short* wvT   = (unsigned short*)(ws + off);  off += szWT;
    unsigned short* wrT   = (unsigned short*)(ws + off);  off += szWT;
    unsigned short* woT   = (unsigned short*)(ws + off);  off += szWT;
    float*          An    = (float*)(ws + off);           off += 4096;
    float*          dt    = (float*)(ws + off);           off += (size_t)Mrows * 4;
    float*          Bx    = (float*)(ws + off);           off += (size_t)Mrows * Sd * 4;
    float*          states= (float*)(ws + off);           off += (size_t)Mrows * Sd * 4;
    unsigned short* kbuf  = (unsigned short*)(ws + off);  off += szXR;
    unsigned short* vbuf  = (unsigned short*)(ws + off);  off += szXR;
    unsigned short* rbuf  = (unsigned short*)(ws + off);  off += szXR;
    unsigned short* ysbuf = (unsigned short*)(ws + off);  off += szXR;
    float*          rwkvo = (float*)(ws + off);           off += szXS;

    float* outY   = (float*)d_out;                               // (B,L,D)
    float* outSSM = outY + (size_t)Mrows * Dm;                   // (B,S)
    float* outWKV = outSSM + (size_t)Bsz * Sd;                   // (B,H,HD,HD)

    // 1) dual LN
    ln_dual_kernel<<<Mrows, 256, 0, stream>>>(x, g_ssm, b_ssm, g_rwkv, b_rwkv, xs, xr);
    // 2) weight transpose + bf16
    dim3 tg(Dm / 16, Dm / 16);
    transpose_bf16_kernel<<<tg, 256, 0, stream>>>(wk, wkT);
    transpose_bf16_kernel<<<tg, 256, 0, stream>>>(wv, wvT);
    transpose_bf16_kernel<<<tg, 256, 0, stream>>>(wr, wrT);
    transpose_bf16_kernel<<<tg, 256, 0, stream>>>(wo, woT);
    // 3) spectral-normalized A
    spectral_an_kernel<<<1, 256, 0, stream>>>(A, An);
    // 4) dt + Bx
    dtbx_kernel<<<Mrows, 256, 0, stream>>>(xs, B_w, dt_w, dt_b, dt, Bx);
    // 5) SSM scan (writes final state into d_out)
    ssm_scan_kernel<<<1, 128, 0, stream>>>(An, dt, Bx, sn_g, sn_b, states, outSSM);
    // 6) ssm_out (in-place over xs)
    ssm_out_kernel<<<Mrows, 256, 0, stream>>>(states, C_w, D_vec, xs);
    // 7) k / v / r projections via WMMA (bf16 outputs)
    dim3 gg(Mrows / 16, Dm / 512);
    gemm_bf16_wmma_kernel<<<gg, 256, 0, stream>>>(xr, wkT, kbuf, Mrows, Dm, Dm, 1);
    gemm_bf16_wmma_kernel<<<gg, 256, 0, stream>>>(xr, wvT, vbuf, Mrows, Dm, Dm, 1);
    gemm_bf16_wmma_kernel<<<gg, 256, 0, stream>>>(xr, wrT, rbuf, Mrows, Dm, Dm, 1);
    // 8) RWKV scan (writes final wkv state into d_out)
    rwkv_scan_kernel<<<Bsz * Hn, 256, 0, stream>>>(kbuf, vbuf, rbuf, time_decay, time_first,
                                                   ysbuf, outWKV);
    // 9) output projection via WMMA (f32 output)
    gemm_bf16_wmma_kernel<<<gg, 256, 0, stream>>>(ysbuf, woT, rwkvo, Mrows, Dm, Dm, 0);
    // 10) gate-combine + LN + residual
    combine_kernel<<<Mrows, 256, 0, stream>>>(x, xs, rwkvo, g_out, b_out, outY);
}
